// GEMConv_50972671869208
// MI455X (gfx1250) — matrix operand order repaired and checked by reading, not compiled, not run on hardware
//
#include <hip/hip_runtime.h>
#include <hip/hip_bf16.h>

typedef __attribute__((ext_vector_type(16))) __bf16 v16bf;
typedef __attribute__((ext_vector_type(8)))  __bf16 v8bf;
typedef __attribute__((ext_vector_type(4)))  __bf16 v4bf;
typedef __attribute__((ext_vector_type(8)))  float  v8f;

#define H 256

// ---------------------------------------------------------------------------
// WMMA helpers (CDNA5 V_WMMA_F32_16X16X32_BF16, layouts per ISA 7.12.2)
// ---------------------------------------------------------------------------
__device__ __forceinline__ v8f wmma_bf16(v16bf a, v16bf b, v8f c) {
    return __builtin_amdgcn_wmma_f32_16x16x32_bf16(
        /*neg_a=*/false, a, /*neg_b=*/false, b,
        /*c_mod=*/(short)0, c, /*reuse_a=*/false, /*reuse_b=*/false);
}

// A fragment from LDS: 16x32 bf16 row-major (stride in elements).
// Lane L(0-15): M=L, K={0..7,16..23}; Lane L(16-31): M=L-16, K={8..15,24..31}
// Emitted as two 16-byte ds loads.
__device__ __forceinline__ v16bf load_a_lds(const __bf16* base, int stride,
                                            int lane, int k0) {
    int m  = lane & 15;
    int lo = (lane < 16) ? 0 : 8;
    const __bf16* p = base + m * stride + k0 + lo;
    v8bf a0 = *(const v8bf*)p;          // K = lo+0..lo+7
    v8bf a1 = *(const v8bf*)(p + 16);   // K = lo+16..lo+23
    return __builtin_shufflevector(a0, a1, 0, 1, 2, 3, 4, 5, 6, 7,
                                           8, 9, 10, 11, 12, 13, 14, 15);
}

// B fragment from fragment-swizzled weights: one 32-byte contiguous load/lane.
__device__ __forceinline__ v16bf load_b_swz(const __bf16* wswz, int tile,
                                            int lane) {
    return *(const v16bf*)&wswz[((size_t)tile * 32 + lane) * 16];
}

// ---------------------------------------------------------------------------
// Weight conversion: f32 row-major W[K][N] -> bf16 WMMA-fragment-swizzled.
// Tile (kb,nt) covers K[kb*32, kb*32+32) x N[nt*16, nt*16+16).
// out[((kb*(N/16)+nt)*32 + l)*16 + i] = W[kb*32 + ((l<16)?0:8) + (i&7) + ((i&8)?16:0)][nt*16 + (l&15)]
// ---------------------------------------------------------------------------
__global__ void swizzle_w_kernel(const float* __restrict__ in,
                                 __bf16* __restrict__ out, int k_dim, int n_dim) {
    int idx = blockIdx.x * blockDim.x + threadIdx.x;
    int total = k_dim * n_dim;
    int stride = gridDim.x * blockDim.x;
    int ntiles = n_dim >> 4;
    for (; idx < total; idx += stride) {
        int i    = idx & 15;
        int l    = (idx >> 4) & 31;
        int tile = idx >> 9;
        int nt   = tile % ntiles;
        int kb   = tile / ntiles;
        int k = kb * 32 + ((l < 16) ? 0 : 8) + (i & 7) + ((i & 8) ? 16 : 0);
        int n = nt * 16 + (l & 15);
        out[idx] = (__bf16)in[(size_t)k * n_dim + n];
    }
}

// ---------------------------------------------------------------------------
// Small utility kernels
// ---------------------------------------------------------------------------
__global__ void zero_f32_kernel(float* __restrict__ p, long long n) {
    long long i = (long long)blockIdx.x * blockDim.x + threadIdx.x;
    long long stride = (long long)gridDim.x * blockDim.x;
    for (; i < n; i += stride) p[i] = 0.0f;
}

__global__ void count_kernel(const int* __restrict__ gid,
                             float* __restrict__ cnt, int n) {
    int i = blockIdx.x * blockDim.x + threadIdx.x;
    if (i < n) atomicAdd(&cnt[gid[i]], 1.0f);
}

__global__ void finalize_cnt_kernel(const float* __restrict__ cnt,
                                    float* __restrict__ rs,
                                    float* __restrict__ inv, int g) {
    int i = blockIdx.x * blockDim.x + threadIdx.x;
    if (i < g) {
        float c = fmaxf(cnt[i], 1.0f);
        rs[i]  = rsqrtf(c);
        inv[i] = 1.0f / c;
    }
}

// agg[dst[e]] += feat[src[e]]   (rows of H floats, float4-vectorized)
__global__ void scatter_gather_kernel(const float* __restrict__ feat,
                                      const int* __restrict__ src,
                                      const int* __restrict__ dst,
                                      float* __restrict__ agg, int ne) {
    long long idx = (long long)blockIdx.x * blockDim.x + threadIdx.x;
    long long total = (long long)ne * (H / 4);
    if (idx >= total) return;
    int e  = (int)(idx >> 6);
    int j4 = ((int)idx & 63) * 4;
    const float4 v = *(const float4*)&feat[(size_t)src[e] * H + j4];
    float* p = &agg[(size_t)dst[e] * H + j4];
    atomicAdd(p + 0, v.x); atomicAdd(p + 1, v.y);
    atomicAdd(p + 2, v.z); atomicAdd(p + 3, v.w);
}

// agg[dst[e]] += nodef[src[e]] + edgef[e]
__global__ void scatter_gather2_kernel(const float* __restrict__ nodef,
                                       const float* __restrict__ edgef,
                                       const int* __restrict__ src,
                                       const int* __restrict__ dst,
                                       float* __restrict__ agg, int ne) {
    long long idx = (long long)blockIdx.x * blockDim.x + threadIdx.x;
    long long total = (long long)ne * (H / 4);
    if (idx >= total) return;
    int e  = (int)(idx >> 6);
    int j4 = ((int)idx & 63) * 4;
    const float4 a = *(const float4*)&nodef[(size_t)src[e] * H + j4];
    const float4 b = *(const float4*)&edgef[(size_t)e * H + j4];
    float* p = &agg[(size_t)dst[e] * H + j4];
    atomicAdd(p + 0, a.x + b.x); atomicAdd(p + 1, a.y + b.y);
    atomicAdd(p + 2, a.z + b.z); atomicAdd(p + 3, a.w + b.w);
}

// graph_repr[gid[n]] += node_out[n] * inv_cnt[gid[n]]
__global__ void pool_kernel(const float* __restrict__ node_out,
                            const int* __restrict__ gid,
                            const float* __restrict__ inv,
                            float* __restrict__ graph_repr, int n) {
    long long idx = (long long)blockIdx.x * blockDim.x + threadIdx.x;
    long long total = (long long)n * (H / 4);
    if (idx >= total) return;
    int e  = (int)(idx >> 6);
    int j4 = ((int)idx & 63) * 4;
    int g  = gid[e];
    float s = inv[g];
    const float4 v = *(const float4*)&node_out[(size_t)e * H + j4];
    float* p = &graph_repr[(size_t)g * H + j4];
    atomicAdd(p + 0, v.x * s); atomicAdd(p + 1, v.y * s);
    atomicAdd(p + 2, v.z * s); atomicAdd(p + 3, v.w * s);
}

// ---------------------------------------------------------------------------
// Fused RBF encode + (rbf @ rbf_w + rbf_b) + scatter-add into agg1.
// One wave = 16 angles; one WMMA per 16-col tile (K=32 == #centers).
// Branch-free scatter: invalid rows scatter 0.0 to row 0.
// ---------------------------------------------------------------------------
__global__ __launch_bounds__(256)
void rbf_scatter_kernel(const float* __restrict__ angle,
                        const int* __restrict__ ba_dst,
                        const __bf16* __restrict__ rbf_wswz,  // swizzled [32,256]
                        const float* __restrict__ rbf_b,      // [256]
                        float* __restrict__ agg1, int a_total) {
    const int tid  = threadIdx.x;
    const int wave = tid >> 5;
    const int lane = tid & 31;
    const int a0   = (blockIdx.x * 8 + wave) * 16;
    if (a0 >= a_total) return;          // wave-uniform: EXEC stays all-ones

    // A fragment: rbf values for row M = lane&15
    const int m  = lane & 15;
    const int lo = (lane < 16) ? 0 : 8;
    const int ar = a0 + m;
    const float ang = (ar < a_total) ? angle[ar] : 0.0f;
    v16bf afr;
#pragma unroll
    for (int i = 0; i < 8; ++i) {
        float d0 = ang - 0.1f * (float)(lo + i);
        float d1 = ang - 0.1f * (float)(lo + 16 + i);
        afr[i]     = (__bf16)__expf(-10.0f * d0 * d0);
        afr[8 + i] = (__bf16)__expf(-10.0f * d1 * d1);
    }

    // D layout: VGPR r -> M = r + (lane<16 ? 0 : 8)
    const int mbase = (lane < 16) ? 0 : 8;
    int   dstrow[8];
    float vmask[8];
#pragma unroll
    for (int r = 0; r < 8; ++r) {
        int arr = a0 + mbase + r;
        bool ok = (arr < a_total);
        dstrow[r] = ok ? ba_dst[arr] : 0;
        vmask[r]  = ok ? 1.0f : 0.0f;
    }

#pragma unroll 1
    for (int nt = 0; nt < 16; ++nt) {
        v16bf bfr = load_b_swz(rbf_wswz, nt, lane);
        v8f c = {};
        c = wmma_bf16(afr, bfr, c);
        const int n = nt * 16 + (lane & 15);
        const float bias = rbf_b[n];
#pragma unroll
        for (int r = 0; r < 8; ++r) {
            atomicAdd(&agg1[(size_t)dstrow[r] * H + n],
                      (c[r] + bias) * vmask[r]);
        }
    }
}

// ---------------------------------------------------------------------------
// Fused GIN MLP: relu(agg @ w1 + b1) @ w2 + b2 -> LayerNorm -> graph-norm
//                -> + residual.  16 rows per workgroup, 8 waves, bf16 WMMA.
// Weights are fragment-swizzled in global (L2-resident, 32B/lane loads).
// LDS: A tile 8KB + h1 16KB + f32 out 16KB = 40KB.
// ---------------------------------------------------------------------------
__global__ __launch_bounds__(256)
void gin_mlp_kernel(const float* __restrict__ agg,     // [rows,256]
                    const float* __restrict__ resid,   // [rows,256]
                    const __bf16* __restrict__ w1swz,  // swizzled [256,512]
                    const float* __restrict__ b1,      // [512]
                    const __bf16* __restrict__ w2swz,  // swizzled [512,256]
                    const float* __restrict__ b2,      // [256]
                    const float* __restrict__ ln_g,
                    const float* __restrict__ ln_b,
                    const int* __restrict__ gid,       // [rows]
                    const float* __restrict__ grs,     // rsqrt(cnt) per graph
                    float* __restrict__ out, int rows) {
    __shared__ __bf16 lds_a[16 * 256];    // 8 KB  : A tile (bf16)
    __shared__ __bf16 lds_h1[16 * 512];   // 16 KB : hidden activations (bf16)
    __shared__ float  lds_o[16 * 256];    // 16 KB : f32 output tile

    const int tid  = threadIdx.x;
    const int wave = tid >> 5;
    const int lane = tid & 31;
    const int row0 = blockIdx.x * 16;

    // ---- stage A: 16x256 f32 -> bf16 in LDS (float4 in, 4xbf16 out)
    for (int i = tid * 4; i < 16 * 256; i += 256 * 4) {
        int m = i >> 8, j = i & 255;
        float4 v = make_float4(0.f, 0.f, 0.f, 0.f);
        if (row0 + m < rows)
            v = *(const float4*)&agg[(size_t)(row0 + m) * H + j];
        v4bf o; o[0] = (__bf16)v.x; o[1] = (__bf16)v.y;
                o[2] = (__bf16)v.z; o[3] = (__bf16)v.w;
        *(v4bf*)&lds_a[i] = o;
    }
    __syncthreads();

    // ---- GEMM1: [16,256] @ [256,512]; wave owns n in [wave*64, wave*64+64)
    v8f acc1[4] = {};
#pragma unroll
    for (int kb = 0; kb < 8; ++kb) {
        // prefetch next k-block's fragments for this wave (global_prefetch_b8)
        __builtin_prefetch(
            &w1swz[(((size_t)((kb + 1) & 7) * 32 + wave * 4) * 32 + lane) * 16],
            0, 1);
        v16bf afr = load_a_lds(lds_a, 256, lane, kb * 32);
#pragma unroll
        for (int t = 0; t < 4; ++t) {
            v16bf bfr = load_b_swz(w1swz, kb * 32 + wave * 4 + t, lane);
            acc1[t] = wmma_bf16(afr, bfr, acc1[t]);
        }
    }

    // ---- +b1, relu, bf16 -> lds_h1
#pragma unroll
    for (int t = 0; t < 4; ++t) {
        int n = wave * 64 + t * 16 + (lane & 15);
        float bias = b1[n];
#pragma unroll
        for (int r = 0; r < 8; ++r) {
            int m = r + ((lane < 16) ? 0 : 8);
            float v = acc1[t][r] + bias;
            lds_h1[m * 512 + n] = (__bf16)(v > 0.0f ? v : 0.0f);
        }
    }
    __syncthreads();

    // ---- GEMM2: [16,512] @ [512,256]; wave owns n in [wave*32, wave*32+32)
    v8f acc2[2] = {};
#pragma unroll
    for (int kb = 0; kb < 16; ++kb) {
        __builtin_prefetch(
            &w2swz[(((size_t)((kb + 1) & 15) * 16 + wave * 2) * 32 + lane) * 16],
            0, 1);
        v16bf afr = load_a_lds(lds_h1, 512, lane, kb * 32);
#pragma unroll
        for (int t = 0; t < 2; ++t) {
            v16bf bfr = load_b_swz(w2swz, kb * 16 + wave * 2 + t, lane);
            acc2[t] = wmma_bf16(afr, bfr, acc2[t]);
        }
    }

    // ---- +b2 into f32 LDS tile
#pragma unroll
    for (int t = 0; t < 2; ++t) {
        int n = wave * 32 + t * 16 + (lane & 15);
        float bias = b2[n];
#pragma unroll
        for (int r = 0; r < 8; ++r) {
            int m = r + ((lane < 16) ? 0 : 8);
            lds_o[m * 256 + n] = acc2[t][r] + bias;
        }
    }
    __syncthreads();

    // ---- LayerNorm + graph-norm + residual; 2 rows per wave
#pragma unroll
    for (int rr = 0; rr < 2; ++rr) {
        int m = wave * 2 + rr;
        int grow = row0 + m;
        if (grow >= rows) continue;
        float s = 0.0f, s2 = 0.0f;
#pragma unroll
        for (int jj = 0; jj < 8; ++jj) {
            float v = lds_o[m * 256 + jj * 32 + lane];
            s += v; s2 += v * v;
        }
#pragma unroll
        for (int off = 16; off > 0; off >>= 1) {
            s  += __shfl_xor(s, off, 32);
            s2 += __shfl_xor(s2, off, 32);
        }
        float mean = s * (1.0f / 256.0f);
        float var  = s2 * (1.0f / 256.0f) - mean * mean;
        float rstd = rsqrtf(var + 1e-5f);
        float gn   = grs[gid[grow]];
#pragma unroll
        for (int jj = 0; jj < 8; ++jj) {
            int j = jj * 32 + lane;
            float v = (lds_o[m * 256 + j] - mean) * rstd * ln_g[j] + ln_b[j];
            out[(size_t)grow * H + j] = v * gn + resid[(size_t)grow * H + j];
        }
    }
}

// ---------------------------------------------------------------------------
// Host orchestration
// ---------------------------------------------------------------------------
extern "C" void kernel_launch(void* const* d_in, const int* in_sizes, int n_in,
                              void* d_out, int out_size, void* d_ws, size_t ws_size,
                              hipStream_t stream) {
    (void)n_in; (void)ws_size;
    const float* node_hidden = (const float*)d_in[0];
    const float* edge_hidden = (const float*)d_in[1];
    const float* bond_angle  = (const float*)d_in[2];
    const int*   ab_src      = (const int*)d_in[3];
    const int*   ab_dst      = (const int*)d_in[4];
    const int*   ba_src      = (const int*)d_in[5];
    const int*   ba_dst      = (const int*)d_in[6];
    const int*   node_gid    = (const int*)d_in[7];
    const int*   bond_gid    = (const int*)d_in[8];
    // d_in[9] = num_graphs scalar (G derived from out_size instead)
    const float* rbf_w  = (const float*)d_in[10];
    const float* rbf_b  = (const float*)d_in[11];
    const float* ba_w1  = (const float*)d_in[12];
    const float* ba_b1  = (const float*)d_in[13];
    const float* ba_w2  = (const float*)d_in[14];
    const float* ba_b2  = (const float*)d_in[15];
    const float* ba_lng = (const float*)d_in[16];
    const float* ba_lnb = (const float*)d_in[17];
    const float* ab_w1  = (const float*)d_in[18];
    const float* ab_b1  = (const float*)d_in[19];
    const float* ab_w2  = (const float*)d_in[20];
    const float* ab_b2  = (const float*)d_in[21];
    const float* ab_lng = (const float*)d_in[22];
    const float* ab_lnb = (const float*)d_in[23];

    const int N = in_sizes[0] / H;
    const int E = in_sizes[1] / H;
    const int A = in_sizes[2];
    const int G = out_size / H - N - E;

    float* out_node  = (float*)d_out;
    float* out_edge  = out_node + (size_t)N * H;
    float* out_graph = out_edge + (size_t)E * H;

    // workspace layout
    char* ws = (char*)d_ws;
    size_t off = 0;
    auto walloc = [&](size_t bytes) -> void* {
        void* p = ws + off;
        off = (off + bytes + 255) & ~(size_t)255;
        return p;
    };
    float*  agg1     = (float*)walloc((size_t)E * H * 4);    // 128 MB (L2-resident)
    float*  agg2     = (float*)walloc((size_t)N * H * 4);    //  64 MB (L2-resident)
    __bf16* ba_w1b   = (__bf16*)walloc((size_t)256 * 512 * 2);
    __bf16* ba_w2b   = (__bf16*)walloc((size_t)512 * 256 * 2);
    __bf16* ab_w1b   = (__bf16*)walloc((size_t)256 * 512 * 2);
    __bf16* ab_w2b   = (__bf16*)walloc((size_t)512 * 256 * 2);
    __bf16* rbf_wb   = (__bf16*)walloc((size_t)32 * 256 * 2);
    float*  bond_cnt = (float*)walloc((size_t)G * 4);
    float*  node_cnt = (float*)walloc((size_t)G * 4);
    float*  bond_rs  = (float*)walloc((size_t)G * 4);
    float*  node_rs  = (float*)walloc((size_t)G * 4);
    float*  node_inv = (float*)walloc((size_t)G * 4);
    float*  dummy    = (float*)walloc((size_t)G * 4);

    // 1) weight conversion to WMMA-fragment-swizzled bf16
    swizzle_w_kernel<<<256, 256, 0, stream>>>(ba_w1, ba_w1b, 256, 512);
    swizzle_w_kernel<<<256, 256, 0, stream>>>(ba_w2, ba_w2b, 512, 256);
    swizzle_w_kernel<<<256, 256, 0, stream>>>(ab_w1, ab_w1b, 256, 512);
    swizzle_w_kernel<<<256, 256, 0, stream>>>(ab_w2, ab_w2b, 512, 256);
    swizzle_w_kernel<<<64, 256, 0, stream>>>(rbf_w, rbf_wb, 32, 256);

    // 2) zero accumulators / counts / graph output
    zero_f32_kernel<<<2048, 256, 0, stream>>>(agg1, (long long)E * H);
    zero_f32_kernel<<<2048, 256, 0, stream>>>(agg2, (long long)N * H);
    zero_f32_kernel<<<64, 256, 0, stream>>>(bond_cnt, G);
    zero_f32_kernel<<<64, 256, 0, stream>>>(node_cnt, G);
    zero_f32_kernel<<<512, 256, 0, stream>>>(out_graph, (long long)G * H);

    // 3) per-graph counts -> rsqrt & reciprocal
    count_kernel<<<(E + 255) / 256, 256, 0, stream>>>(bond_gid, bond_cnt, E);
    count_kernel<<<(N + 255) / 256, 256, 0, stream>>>(node_gid, node_cnt, N);
    finalize_cnt_kernel<<<(G + 255) / 256, 256, 0, stream>>>(bond_cnt, bond_rs, dummy, G);
    finalize_cnt_kernel<<<(G + 255) / 256, 256, 0, stream>>>(node_cnt, node_rs, node_inv, G);

    // 4) bond-angle GNN: scatter edge_hidden[ba_src] + fused RBF matmul -> agg1
    {
        long long total = (long long)A * (H / 4);
        scatter_gather_kernel<<<(unsigned)((total + 255) / 256), 256, 0, stream>>>(
            edge_hidden, ba_src, ba_dst, agg1, A);
        rbf_scatter_kernel<<<(A + 127) / 128, 256, 0, stream>>>(
            bond_angle, ba_dst, rbf_wb, rbf_b, agg1, A);
    }

    // 5) bond-angle GNN MLP -> edge_out
    gin_mlp_kernel<<<(E + 15) / 16, 256, 0, stream>>>(
        agg1, edge_hidden, ba_w1b, ba_b1, ba_w2b, ba_b2,
        ba_lng, ba_lnb, bond_gid, bond_rs, out_edge, E);

    // 6) atom-bond GNN: scatter node_hidden[ab_src] + edge_out -> agg2
    {
        long long total = (long long)E * (H / 4);
        scatter_gather2_kernel<<<(unsigned)((total + 255) / 256), 256, 0, stream>>>(
            node_hidden, out_edge, ab_src, ab_dst, agg2, E);
    }

    // 7) atom-bond GNN MLP -> node_out
    gin_mlp_kernel<<<(N + 15) / 16, 256, 0, stream>>>(
        agg2, node_hidden, ab_w1b, ab_b1, ab_w2b, ab_b2,
        ab_lng, ab_lnb, node_gid, node_rs, out_node, N);

    // 8) mean pooling into graph_repr
    {
        long long total = (long long)N * (H / 4);
        pool_kernel<<<(unsigned)((total + 255) / 256), 256, 0, stream>>>(
            out_node, node_gid, node_inv, out_graph, N);
    }
}
